// MultichannelMultiheadAttention_47588237640245
// MI455X (gfx1250) — matrix operand order repaired and checked by reading, not compiled, not run on hardware
//
#include <hip/hip_runtime.h>
#include <hip/hip_bf16.h>

// ---------------- problem constants ----------------
constexpr int B_  = 2;
constexpr int C_  = 8;
constexpr int H_  = 512;
constexpr int W_  = 1024;
constexpr int NH_ = 8;
constexpr int DH_ = H_ / NH_;          // 64
constexpr float QK_SCALE = 0.044194173824159216f;  // 1/sqrt(512)

typedef __attribute__((ext_vector_type(16))) __bf16    v16bf;
typedef __attribute__((ext_vector_type(8)))  float     v8f;
typedef __attribute__((ext_vector_type(4)))  unsigned  v4u;
typedef __attribute__((ext_vector_type(8)))  unsigned  v8u;

union V16U { v16bf v; uint4 q[2]; };

// ---------------- bf16 helpers (bit-level) ----------------
__device__ __forceinline__ unsigned short f2bf(float f) {
    unsigned int u = __float_as_uint(f);
    u += 0x7FFFu + ((u >> 16) & 1u);     // round-to-nearest-even
    return (unsigned short)(u >> 16);
}
__device__ __forceinline__ float bf2f(unsigned short h) {
    return __uint_as_float(((unsigned int)h) << 16);
}
__device__ __forceinline__ unsigned lds_off(const void* p) {
    // generic LDS pointer: addr[31:0] is the LDS byte offset (ISA 10.2 aperture map)
    return (unsigned)(unsigned long long)p;
}

// ---------------- CDNA5 async / TDM data movement ----------------
__device__ __forceinline__ void async_ld_b128(unsigned lds_addr, const void* base,
                                              unsigned byte_off) {
    asm volatile("global_load_async_to_lds_b128 %0, %1, %2"
                 :: "v"(lds_addr), "v"(byte_off), "s"(base) : "memory");
}
__device__ __forceinline__ void wait_async0() {
    asm volatile("s_wait_asynccnt 0x0" ::: "memory");
}

// Build + issue a 2D-tile TDM descriptor (2-byte elements).
// pad=true inserts 16B of LDS padding after every 128B row (-> 144B LDS pitch).
__device__ __forceinline__ void tdm_2d(bool is_store, bool pad,
                                       unsigned lds_addr, const void* gptr,
                                       unsigned kelems, unsigned rows,
                                       unsigned stride_elems) {
    unsigned long long ga = (unsigned long long)gptr;
    v4u g0;
    g0[0] = 1u;                                        // count=1 (valid user D#)
    g0[1] = lds_addr;
    g0[2] = (unsigned)(ga & 0xFFFFFFFFu);              // global_addr[31:0]
    g0[3] = (unsigned)((ga >> 32) & 0x01FFFFFFu) | (2u << 30);  // addr[56:32]|type=2
    const unsigned td0 = 0x7FFFFFFFu, td1 = 0x7FFFFFFFu;        // no OOB clipping
    v8u g1;
    g1[0] = (1u << 16)                                 // data_size = 2B
          | (pad ? ((1u << 20) | (4u << 22) | (3u << 25)) : 0u);
    g1[1] = (td0 & 0xFFFFu) << 16;                     // barrier_addr=0 | td0[15:0]
    g1[2] = (td0 >> 16) | ((td1 & 0xFFFFu) << 16);
    g1[3] = (td1 >> 16) | (kelems << 16);              // tile_dim0
    g1[4] = rows;                                      // tile_dim1 | tile_dim2=0
    g1[5] = stride_elems;                              // tensor_dim0_stride[31:0]
    g1[6] = 0u;
    g1[7] = 0u;
    v4u g2 = {0u, 0u, 0u, 0u};
    v4u g3 = {0u, 0u, 0u, 0u};
    if (is_store)
        asm volatile("tensor_store_from_lds %0, %1, %2, %3"
                     :: "s"(g0), "s"(g1), "s"(g2), "s"(g3) : "memory");
    else
        asm volatile("tensor_load_to_lds %0, %1, %2, %3"
                     :: "s"(g0), "s"(g1), "s"(g2), "s"(g3) : "memory");
}

// ---------------- WMMA fragment loads (CDNA5 ISA layouts) -------
__device__ __forceinline__ v16bf load_frag_a(const unsigned short* tile, int ld,
                                             int row0, int ks, int lane) {
    int r  = row0 + (lane & 15);
    int kb = ks + ((lane < 16) ? 0 : 8);
    const unsigned short* p = tile + r * ld + kb;
    V16U u;
    u.q[0] = *(const uint4*)(p);
    u.q[1] = *(const uint4*)(p + 16);
    return u.v;
}
__device__ __forceinline__ v16bf load_frag_b(const unsigned short* tileT, int ld,
                                             int col0, int ks, int lane) {
    int c  = col0 + (lane & 15);
    int kb = ks + ((lane < 16) ? 0 : 8);
    const unsigned short* p = tileT + c * ld + kb;
    V16U u;
    u.q[0] = *(const uint4*)(p);
    u.q[1] = *(const uint4*)(p + 16);
    return u.v;
}
__device__ __forceinline__ v8f wmma_bf16(v16bf a, v16bf b, v8f c) {
    return __builtin_amdgcn_wmma_f32_16x16x32_bf16(false, a, false, b,
                                                   (short)0, c, false, false);
}

// ---------------- kernels ----------------

__global__ __launch_bounds__(256)
void cvt_bf16_kernel(const float* __restrict__ s, unsigned short* __restrict__ d,
                     long long n) {
    long long i      = (long long)blockIdx.x * blockDim.x + threadIdx.x;
    long long stride = (long long)gridDim.x * blockDim.x;
    for (; i < n; i += stride) d[i] = f2bf(s[i]);
}

// x (bc,h,w) fp32 -> xT (bc,w,h) bf16, LDS-tiled 32x32 transpose.
__global__ __launch_bounds__(256)
void transpose_cvt_kernel(const float* __restrict__ src,
                          unsigned short* __restrict__ dst) {
    __shared__ float tile[32][33];
    int bc = blockIdx.z;
    int h0 = blockIdx.y * 32, w0 = blockIdx.x * 32;
    int tx = threadIdx.x & 31, ty = threadIdx.x >> 5;   // 32 x 8
    const float* s = src + ((size_t)bc * H_ + h0) * W_ + w0;
#pragma unroll
    for (int i = 0; i < 32; i += 8)
        tile[ty + i][tx] = s[(size_t)(ty + i) * W_ + tx];
    __syncthreads();
    unsigned short* d = dst + ((size_t)bc * W_ + w0) * H_ + h0;
#pragma unroll
    for (int i = 0; i < 32; i += 8)
        d[(size_t)(ty + i) * H_ + tx] = f2bf(tile[tx][ty + i]);
}

// Batched bf16 GEMM: C[z] = A[z%aMod] (MxK, row-major) * B[z] (KxN, K-major).
// WG tile 128x64, wave tile 32x32, K-step 64 (8 wmma / stage).
// Double-buffered LDS; A tile TDM by wave 0, B tile TDM by wave 1, so the
// next stage is in flight while the current one is in the WMMA pipe.
__global__ __launch_bounds__(256)
void gemm_bf16_kernel(const unsigned short* __restrict__ A,
                      const unsigned short* __restrict__ Bm,
                      float* Cf, unsigned short* Cb,
                      int M, int N, int K, int lda, int ldbT, int ldc,
                      long long aStride, int aMod,
                      long long bStride, long long cStride, float scale) {
    __shared__ __attribute__((aligned(16))) unsigned short As[2][128][72];
    __shared__ __attribute__((aligned(16))) unsigned short Bt[2][64][72];

    const int z  = blockIdx.z;
    const unsigned short* Ab = A  + (long long)(z % aMod) * aStride;
    const unsigned short* Bb = Bm + (long long)z * bStride;
    const int m0 = blockIdx.x * 128;
    const int n0 = blockIdx.y * 64;
    const int t    = threadIdx.x;
    const int wave = t >> 5, lane = t & 31;
    const int wm = (wave & 3) * 32;
    const int wn = (wave >> 2) * 32;

    auto stage = [&](int buf, int k0) {
        if (wave == 0)
            tdm_2d(false, true, lds_off(&As[buf][0][0]),
                   Ab + (size_t)m0 * lda + k0, 64u, 128u, (unsigned)lda);
        else if (wave == 1)
            tdm_2d(false, true, lds_off(&Bt[buf][0][0]),
                   Bb + (size_t)n0 * ldbT + k0, 64u, 64u, (unsigned)ldbT);
    };

    v8f acc[2][2] = {};
    const int nsteps = K >> 6;
    stage(0, 0);

    for (int i = 0; i < nsteps; ++i) {
        if (wave < 2) __builtin_amdgcn_s_wait_tensorcnt(0);
        __syncthreads();                       // stage i visible to all waves
        if (i + 1 < nsteps) stage((i + 1) & 1, (i + 1) << 6);  // overlap next
        const unsigned short* Asb = &As[i & 1][0][0];
        const unsigned short* Btb = &Bt[i & 1][0][0];
#pragma unroll
        for (int ks = 0; ks < 64; ks += 32) {
            v16bf a0 = load_frag_a(Asb, 72, wm,      ks, lane);
            v16bf a1 = load_frag_a(Asb, 72, wm + 16, ks, lane);
            v16bf b0 = load_frag_b(Btb, 72, wn,      ks, lane);
            v16bf b1 = load_frag_b(Btb, 72, wn + 16, ks, lane);
            acc[0][0] = wmma_bf16(a0, b0, acc[0][0]);
            acc[0][1] = wmma_bf16(a0, b1, acc[0][1]);
            acc[1][0] = wmma_bf16(a1, b0, acc[1][0]);
            acc[1][1] = wmma_bf16(a1, b1, acc[1][1]);
        }
        __syncthreads();                       // all reads done before re-stage
    }

    float*          Co  = Cf ? Cf + (long long)z * cStride : nullptr;
    unsigned short* Cbo = Cb ? Cb + (long long)z * cStride : nullptr;
#pragma unroll
    for (int fm = 0; fm < 2; ++fm)
#pragma unroll
        for (int fn = 0; fn < 2; ++fn) {
            int mbase = m0 + wm + fm * 16 + ((lane >> 4) ? 8 : 0);
            int col   = n0 + wn + fn * 16 + (lane & 15);
#pragma unroll
            for (int r = 0; r < 8; ++r) {
                float v = acc[fm][fn][r] * scale;
                size_t idx = (size_t)(mbase + r) * ldc + col;
                if (Cf) Co[idx] = v;
                else    Cbo[idx] = f2bf(v);
            }
        }
}

// conv(1x3 over channels) + bias + head split (+ RoPE for q/k).
// mode 0: Q -> [bcn][pos][d] ; mode 1: K -> [bcn][pos][d] ; mode 2: V -> [bcn][d][pos]
__global__ __launch_bounds__(256)
void conv_rope_kernel(const unsigned short* __restrict__ lin,
                      const float* __restrict__ kw, const float* __restrict__ bias,
                      unsigned short* __restrict__ out, int mode) {
    long long idx = (long long)blockIdx.x * blockDim.x + threadIdx.x;
    int w  = (int)(idx % W_);
    int hp = (int)((idx / W_) % (H_ / 2));
    int b  = (int)(idx / ((long long)W_ * (H_ / 2)));
    int h0 = hp * 2;

    float vin[2][C_][3];
#pragma unroll
    for (int hh = 0; hh < 2; ++hh)
#pragma unroll
        for (int ci = 0; ci < C_; ++ci) {
            const unsigned short* p =
                lin + ((size_t)(b * C_ + ci) * H_ + (h0 + hh)) * W_ + w;
            vin[hh][ci][0] = (w > 0)      ? bf2f(p[-1]) : 0.f;
            vin[hh][ci][1] =                bf2f(p[0]);
            vin[hh][ci][2] = (w < W_ - 1) ? bf2f(p[1])  : 0.f;
        }

    float y[2][C_];
#pragma unroll
    for (int co = 0; co < C_; ++co) {
        float s0 = bias[co], s1 = bias[co];
#pragma unroll
        for (int ci = 0; ci < C_; ++ci) {
            const float* kp = kw + (co * C_ + ci) * 3;
#pragma unroll
            for (int tp = 0; tp < 3; ++tp) {
                s0 = fmaf(kp[tp], vin[0][ci][tp], s0);
                s1 = fmaf(kp[tp], vin[1][ci][tp], s1);
            }
        }
        y[0][co] = s0; y[1][co] = s1;
    }

    if (mode != 2) {                      // RoPE over seq (w); pair (d, d+1) local
        int d = h0 & (DH_ - 1);           // even
        float ang = (float)w * powf(10000.f, -(float)d / (float)DH_);
        float cs = cosf(ang), sn = sinf(ang);
#pragma unroll
        for (int co = 0; co < C_; ++co) {
            float e = y[0][co], o = y[1][co];
            y[0][co] = e * cs - o * sn;
            y[1][co] = o * cs + e * sn;
        }
    }

#pragma unroll
    for (int hh = 0; hh < 2; ++hh)
#pragma unroll
        for (int co = 0; co < C_; ++co) {
            int h = h0 + hh, n = h >> 6, d = h & (DH_ - 1);
            size_t o;
            if (mode == 2)   // V transposed: [bcn][d][pos]
                o = (((size_t)(b * C_ + co) * NH_ + n) * DH_ + d) * W_ + w;
            else             // Q, K: [bcn][pos][d]
                o = (((size_t)(b * C_ + co) * NH_ + n) * W_ + w) * DH_ + d;
            out[o] = f2bf(y[hh][co]);
        }
}

// Per-row softmax statistics (max, sum of exp) over the 1024 key dim.
__global__ __launch_bounds__(256)
void softmax_stats_kernel(const float* __restrict__ S, float2* __restrict__ st) {
    __shared__ float red[256];
    size_t row = blockIdx.x;
    const float* p = S + row * (size_t)W_;
    int t = threadIdx.x;
    float4 v = ((const float4*)p)[t];
    float m = fmaxf(fmaxf(v.x, v.y), fmaxf(v.z, v.w));
    red[t] = m; __syncthreads();
    for (int s = 128; s > 0; s >>= 1) {
        if (t < s) red[t] = fmaxf(red[t], red[t + s]);
        __syncthreads();
    }
    m = red[0]; __syncthreads();
    float sum = expf(v.x - m) + expf(v.y - m) + expf(v.z - m) + expf(v.w - m);
    red[t] = sum; __syncthreads();
    for (int s = 128; s > 0; s >>= 1) {
        if (t < s) red[t] += red[t + s];
        __syncthreads();
    }
    if (t == 0) st[row] = make_float2(m, red[0]);
    // cluster-wide sync path (hardware NOP when ClusterID==0)
    if (__builtin_amdgcn_cluster_id_x() > 0)
        __builtin_amdgcn_s_cluster_barrier();
}

// a = softmax(S) * V. P recomputed on the fly from S + stats; V is [d][pos].
// Double-buffered: V tiles via async-to-LDS, P tiles via exp+ds_store, both
// for stage i+1 issued before stage i's WMMAs. Result tile staged in LDS and
// written out with one tensor_store_from_lds (a^T layout, row stride H).
__global__ __launch_bounds__(256)
void pv_kernel(const float* __restrict__ S, const float2* __restrict__ st,
               const unsigned short* __restrict__ VT,
               unsigned short* __restrict__ aT) {
    __shared__ __attribute__((aligned(16))) unsigned short Ps[2][128][72];
    __shared__ __attribute__((aligned(16))) unsigned short Vt[2][64][72];

    const int z  = blockIdx.y;                 // head instance 0..127
    const int q0 = blockIdx.x * 128;
    const float*          Sb  = S  + (size_t)z * W_ * W_;
    const float2*         stb = st + (size_t)z * W_;
    const unsigned short* Vb  = VT + (size_t)z * DH_ * W_;

    const int t = threadIdx.x, wave = t >> 5, lane = t & 31;
    const int wm = (wave & 3) * 32, wn = (wave >> 2) * 32;

    const int pr = t >> 1, pkg = (t & 1) * 32;       // P staging: 32 elems/thread
    const int vn = t >> 2, vkg = (t & 3) * 16;       // V staging: 32B/thread

    auto stageV = [&](int buf, int k0) {
        unsigned voff = (unsigned)(((size_t)vn * W_ + k0 + vkg) * 2u);
        unsigned dst  = lds_off(&Vt[buf][vn][vkg]);
        async_ld_b128(dst,      Vb, voff);
        async_ld_b128(dst + 16, Vb, voff + 16);
    };
    auto stageP = [&](int buf, int k0) {
        int q = q0 + pr;
        float2 ms = stb[q];
        float rinv = 1.f / ms.y;
        const float* sp = Sb + (size_t)q * W_ + k0 + pkg;
        if (k0 + 64 < W_) __builtin_prefetch(sp + 64, 0, 1);
        unsigned short tmp[32];
#pragma unroll
        for (int i = 0; i < 32; ++i)
            tmp[i] = f2bf(expf(sp[i] - ms.x) * rinv);
#pragma unroll
        for (int i = 0; i < 4; ++i)
            *(uint4*)&Ps[buf][pr][pkg + i * 8] = ((uint4*)tmp)[i];
    };

    v8f acc[2][2] = {};
    const int nsteps = W_ >> 6;                 // 16
    stageV(0, 0);
    stageP(0, 0);

    for (int i = 0; i < nsteps; ++i) {
        wait_async0();                          // own V-tile writes done
        __syncthreads();                        // stage i visible to all waves
        if (i + 1 < nsteps) {                   // overlap stage i+1 with compute
            stageV((i + 1) & 1, (i + 1) << 6);
            stageP((i + 1) & 1, (i + 1) << 6);
        }
        const unsigned short* Pb  = &Ps[i & 1][0][0];
        const unsigned short* Vtb = &Vt[i & 1][0][0];
#pragma unroll
        for (int ks = 0; ks < 64; ks += 32) {
            v16bf a0 = load_frag_a(Pb,  72, wm,      ks, lane);
            v16bf a1 = load_frag_a(Pb,  72, wm + 16, ks, lane);
            v16bf b0 = load_frag_b(Vtb, 72, wn,      ks, lane);
            v16bf b1 = load_frag_b(Vtb, 72, wn + 16, ks, lane);
            acc[0][0] = wmma_bf16(a0, b0, acc[0][0]);
            acc[0][1] = wmma_bf16(a0, b1, acc[0][1]);
            acc[1][0] = wmma_bf16(a1, b0, acc[1][0]);
            acc[1][1] = wmma_bf16(a1, b1, acc[1][1]);
        }
        __syncthreads();
    }

    // C tile -> LDS (reuse retired Ps[0] buffer: 128x64 bf16 = 16KB < 18KB)
    unsigned short (*Co)[64] = (unsigned short (*)[64])&Ps[0][0][0];
#pragma unroll
    for (int fm = 0; fm < 2; ++fm)
#pragma unroll
        for (int fn = 0; fn < 2; ++fn) {
            int rbase = wm + fm * 16 + ((lane >> 4) ? 8 : 0);
            int col   = wn + fn * 16 + (lane & 15);
#pragma unroll
            for (int r = 0; r < 8; ++r)
                Co[rbase + r][col] = f2bf(acc[fm][fn][r]);
        }
    __syncthreads();

    // one TDM store: 128 rows x 64 cols into a^T[bc][q][h], row stride H
    const int bc = z >> 3, hn = z & 7;
    if (wave == 0) {
        unsigned short* gdst = aT + (size_t)bc * W_ * H_ + (size_t)q0 * H_ + hn * DH_;
        tdm_2d(true, false, lds_off(&Co[0][0]), gdst, 64u, 128u, (unsigned)H_);
        __builtin_amdgcn_s_wait_tensorcnt(0);
    }
}

// ---------------- workspace layout (bytes) ----------------
constexpr size_t NX  = (size_t)B_ * C_ * H_ * W_;   // 8,388,608
constexpr size_t NW  = (size_t)C_ * H_ * H_;        // 2,097,152
constexpr size_t OFF_XT   = 0;
constexpr size_t OFF_WQB  = OFF_XT   + NX * 2;
constexpr size_t OFF_WKB  = OFF_WQB  + NW * 2;
constexpr size_t OFF_WVB  = OFF_WKB  + NW * 2;
constexpr size_t OFF_WOB  = OFF_WVB  + NW * 2;
constexpr size_t OFF_LINQ = OFF_WOB  + NW * 2;
constexpr size_t OFF_LINK = OFF_LINQ + NX * 2;
constexpr size_t OFF_LINV = OFF_LINK + NX * 2;
constexpr size_t OFF_QH   = OFF_LINV + NX * 2;
constexpr size_t OFF_KH   = OFF_QH   + NX * 2;
constexpr size_t OFF_VT   = OFF_KH   + NX * 2;
constexpr size_t OFF_AT   = OFF_VT   + NX * 2;
constexpr size_t OFF_ST   = OFF_AT   + NX * 2;

extern "C" void kernel_launch(void* const* d_in, const int* in_sizes, int n_in,
                              void* d_out, int out_size, void* d_ws, size_t ws_size,
                              hipStream_t stream) {
    const float* x  = (const float*)d_in[0];
    const float* wq = (const float*)d_in[1];
    const float* kq = (const float*)d_in[2];
    const float* bq = (const float*)d_in[3];
    const float* wk = (const float*)d_in[4];
    const float* kk = (const float*)d_in[5];
    const float* bk = (const float*)d_in[6];
    const float* wv = (const float*)d_in[7];
    const float* kv = (const float*)d_in[8];
    const float* bv = (const float*)d_in[9];
    const float* wo = (const float*)d_in[10];

    float* out = (float*)d_out;
    float* qk  = out + NX;                 // tuple output #2, (b,c,n,q,k)

    char* ws = (char*)d_ws;
    unsigned short* xT   = (unsigned short*)(ws + OFF_XT);
    unsigned short* wqb  = (unsigned short*)(ws + OFF_WQB);
    unsigned short* wkb  = (unsigned short*)(ws + OFF_WKB);
    unsigned short* wvb  = (unsigned short*)(ws + OFF_WVB);
    unsigned short* wob  = (unsigned short*)(ws + OFF_WOB);
    unsigned short* linq = (unsigned short*)(ws + OFF_LINQ);
    unsigned short* link = (unsigned short*)(ws + OFF_LINK);
    unsigned short* linv = (unsigned short*)(ws + OFF_LINV);
    unsigned short* Qh   = (unsigned short*)(ws + OFF_QH);
    unsigned short* Kh   = (unsigned short*)(ws + OFF_KH);
    unsigned short* VT   = (unsigned short*)(ws + OFF_VT);
    unsigned short* aT   = (unsigned short*)(ws + OFF_AT);
    float2*         stp  = (float2*)(ws + OFF_ST);

    // 1) x -> x^T bf16 (K-major B for the projection GEMMs); weights -> bf16
    dim3 gt(W_ / 32, H_ / 32, B_ * C_);
    transpose_cvt_kernel<<<gt, 256, 0, stream>>>(x, xT);
    cvt_bf16_kernel<<<2048, 256, 0, stream>>>(wq, wqb, (long long)NW);
    cvt_bf16_kernel<<<2048, 256, 0, stream>>>(wk, wkb, (long long)NW);
    cvt_bf16_kernel<<<2048, 256, 0, stream>>>(wv, wvb, (long long)NW);
    cvt_bf16_kernel<<<2048, 256, 0, stream>>>(wo, wob, (long long)NW);

    // 2) projections: W[c](512x512) x X[b,c](512x1024), B = x^T (K-major)
    dim3 gp(H_ / 128, W_ / 64, B_ * C_);
    gemm_bf16_kernel<<<gp, 256, 0, stream>>>(wqb, xT, nullptr, linq,
        H_, W_, H_, H_, H_, W_, (long long)H_ * H_, C_,
        (long long)W_ * H_, (long long)H_ * W_, 1.f);
    gemm_bf16_kernel<<<gp, 256, 0, stream>>>(wkb, xT, nullptr, link,
        H_, W_, H_, H_, H_, W_, (long long)H_ * H_, C_,
        (long long)W_ * H_, (long long)H_ * W_, 1.f);
    gemm_bf16_kernel<<<gp, 256, 0, stream>>>(wvb, xT, nullptr, linv,
        H_, W_, H_, H_, H_, W_, (long long)H_ * H_, C_,
        (long long)W_ * H_, (long long)H_ * W_, 1.f);

    // 3) conv + bias + head split (+ RoPE for q,k)
    long long nconv = (long long)B_ * (H_ / 2) * W_;   // 524288
    conv_rope_kernel<<<(unsigned)(nconv / 256), 256, 0, stream>>>(linq, kq, bq, Qh, 0);
    conv_rope_kernel<<<(unsigned)(nconv / 256), 256, 0, stream>>>(link, kk, bk, Kh, 1);
    conv_rope_kernel<<<(unsigned)(nconv / 256), 256, 0, stream>>>(linv, kv, bv, VT, 2);

    // 4) S = Q x K^T / sqrt(512) -> d_out (qk); B = K[pos][d] is already K-major
    dim3 gq(W_ / 128, W_ / 64, B_ * C_ * NH_);
    gemm_bf16_kernel<<<gq, 256, 0, stream>>>(Qh, Kh, qk, nullptr,
        W_, W_, DH_, DH_, DH_, W_, (long long)W_ * DH_, B_ * C_ * NH_,
        (long long)W_ * DH_, (long long)W_ * W_, QK_SCALE);

    // 5) per-row softmax stats
    softmax_stats_kernel<<<(unsigned)(B_ * C_ * NH_ * W_), 256, 0, stream>>>(qk, stp);

    // 6) a = softmax(S) x V -> a^T via tensor_store_from_lds
    dim3 gv(W_ / 128, B_ * C_ * NH_);
    pv_kernel<<<gv, 256, 0, stream>>>(qk, stp, VT, aT);

    // 7) out = wo x a, B = a^T (K-major)
    gemm_bf16_kernel<<<gp, 256, 0, stream>>>(wob, aT, out, nullptr,
        H_, W_, H_, H_, H_, W_, (long long)H_ * H_, C_,
        (long long)W_ * H_, (long long)H_ * W_, 1.f);
}
